// SSSDSAImputer_87230785782055
// MI455X (gfx1250) — matrix operand order repaired and checked by reading, not compiled, not run on hardware
//
#include <hip/hip_runtime.h>
#include <cstddef>

// ---------------------------------------------------------------------------
// SSSD-S4 imputer forward for MI455X (gfx1250).
//  * All 1x1 convs / dense layers -> v_wmma_f32_16x16x32_f16 (fp32 accum).
//  * B-operand tile stored K-contiguous in LDS -> ds_load_b128 fragment loads.
//  * W tile staged to LDS via global_load_async_to_lds_b128 (ASYNCcnt) when
//    available/aligned; cooperative fallback otherwise.
//  * S4 FFT-conv replaced by equivalent bidirectional complex state scan,
//    one conjugate-pair state per wave32 lane, __shfl_xor wave reduction.
// ---------------------------------------------------------------------------

#define SEQ     8192
#define BATCH_N 2

typedef __attribute__((ext_vector_type(16))) _Float16 v16h;
typedef __attribute__((ext_vector_type(8)))  _Float16 v8h;
typedef __attribute__((ext_vector_type(8)))  float    v8f;

#if defined(__HIP_DEVICE_COMPILE__) && __has_builtin(__builtin_amdgcn_global_load_async_to_lds_b128) && __has_builtin(__builtin_amdgcn_s_wait_asynccnt)
#define ASYNC_LDS_OK 1
typedef int v4i_g __attribute__((vector_size(16)));   // matches builtin param type
__device__ __forceinline__ void async_g2l_b128(const void* g, void* l) {
  __builtin_amdgcn_global_load_async_to_lds_b128((v4i_g*)g, (v4i_g*)l, 0, 0);
}
__device__ __forceinline__ void async_wait0() {
  __builtin_amdgcn_s_wait_asynccnt(0);
}
#else
#define ASYNC_LDS_OK 0
#endif

__device__ __forceinline__ float gelu_f(float x) {
  float u = 0.7978845608028654f * (x + 0.044715f * x * x * x);
  return 0.5f * x * (1.0f + tanhf(u));
}
__device__ __forceinline__ float silu_f(float x) { return x / (1.0f + __expf(-x)); }
__device__ __forceinline__ float sigm_f(float x) { return 1.0f / (1.0f + __expf(-x)); }

// ---------------- WMMA GEMM:  Y[M,N] = act(W[M,K] @ X[K,N] + bias) ----------
// Row-major. grid.z = batch. Block = 256 threads = 8 waves.
// Tile 64(M) x 128(N), K panel 32. Wave (wm,wn): wm -> 16-row subtile,
// wn -> 64-col half; 4 accumulators (16x16 f32) per wave.
template<int ACT>  // 0=none 1=relu 2=gelu
__global__ __launch_bounds__(256) void gemm_wmma_f16(
    float* __restrict__ Y, const float* __restrict__ X,
    const float* __restrict__ W, const float* __restrict__ bias,
    int M, int N, int K)
{
  constexpr int MT = 64, NT = 128, KT = 32;
  constexpr int WP = KT + 4;   // fp32 W row pad: 36 dwords = 144B (16B aligned)
  constexpr int XP = KT + 8;   // f16  X row pad: 40 halves = 80B  (16B aligned)
  __shared__ __attribute__((aligned(16))) float    sWf[MT][WP];
  __shared__ __attribute__((aligned(16))) _Float16 sXT[NT][XP];   // [n][k]

  const int bz = blockIdx.z;
  X += (size_t)bz * K * N;
  Y += (size_t)bz * M * N;
  const int m0   = blockIdx.y * MT;
  const int n0   = blockIdx.x * NT;
  const int tid  = threadIdx.x;
  const int lane = tid & 31;
  const int wave = tid >> 5;
  const int wm = wave & 3;
  const int wn = wave >> 2;

#if ASYNC_LDS_OK
  const bool can_async = ((K & 31) == 0) && ((((size_t)W) & 15) == 0) &&
                         (m0 + MT <= M);
#endif

  v8f acc[4];
#pragma unroll
  for (int t = 0; t < 4; ++t)
#pragma unroll
    for (int r = 0; r < 8; ++r) acc[t][r] = 0.0f;

  for (int k0 = 0; k0 < K; k0 += KT) {
    // ---- stage W panel (fp32) -------------------------------------------
#if ASYNC_LDS_OK
    if (can_async) {
      // 64 rows x 32 floats = 512 chunks of 16B; 2 chunks per lane, DMA'd
      // straight into LDS (tracked by ASYNCcnt, overlaps the X transpose).
      for (int i = tid; i < MT * (KT / 4); i += 256) {
        int mm = i >> 3, c4 = i & 7;
        async_g2l_b128(W + (size_t)(m0 + mm) * K + k0 + c4 * 4, &sWf[mm][c4 * 4]);
      }
    } else
#endif
    {
      for (int i = tid; i < MT * KT; i += 256) {
        int mm = i >> 5, kk = i & (KT - 1);
        sWf[mm][kk] = (m0 + mm < M && k0 + kk < K)
                        ? W[(size_t)(m0 + mm) * K + (k0 + kk)] : 0.0f;
      }
    }
    // ---- stage X panel transposed to f16 (K-contiguous per column) ------
    for (int i = tid; i < (KT / 2) * NT; i += 256) {
      int kp = i >> 7, nn = i & (NT - 1);   // consecutive tid -> consecutive nn
      int ka = k0 + 2 * kp, n = n0 + nn;
      float f0 = (ka     < K && n < N) ? X[(size_t)ka * N + n]       : 0.0f;
      float f1 = (ka + 1 < K && n < N) ? X[(size_t)(ka + 1) * N + n] : 0.0f;
      union { _Float16 h[2]; unsigned u; } pk;
      pk.h[0] = (_Float16)f0; pk.h[1] = (_Float16)f1;
      *(unsigned*)&sXT[nn][2 * kp] = pk.u;
    }
#if ASYNC_LDS_OK
    if (can_async) async_wait0();
#endif
    __syncthreads();

    // ---- A fragment 16x32 f16: lane m = lane&15, half = lane>>4,
    // elems 0..15 -> K = 16*half+j (matches B fragment).
    v16h afrag;
    {
      const int m  = wm * 16 + (lane & 15);
      const int kh = (lane >> 4) * 16;
#pragma unroll
      for (int j = 0; j < 8; ++j) afrag[j]     = (_Float16)sWf[m][kh + j];
#pragma unroll
      for (int j = 0; j < 8; ++j) afrag[8 + j] = (_Float16)sWf[m][kh + 8 + j];
    }
    // ---- B fragment 32x16 f16: lane col = lane&15; lanes 0-15 hold K=0..15,
    // lanes 16-31 hold K=16..31 -> two aligned b128 LDS loads per fragment.
#pragma unroll
    for (int t = 0; t < 4; ++t) {
      const int n  = wn * 64 + t * 16 + (lane & 15);
      const int kh = (lane >> 4) * 16;
      v8h b0 = *(const v8h*)&sXT[n][kh];
      v8h b1 = *(const v8h*)&sXT[n][kh + 8];
      v16h bfrag = __builtin_shufflevector(b0, b1,
          0,1,2,3,4,5,6,7,8,9,10,11,12,13,14,15);
      acc[t] = __builtin_amdgcn_wmma_f32_16x16x32_f16(
          false, afrag, false, bfrag, (short)0, acc[t], false, false);
    }
    __syncthreads();
  }

  // C/D layout: VGPR r -> m = 8*(lane>>4)+r, n = lane&15.
  const int nl = lane & 15, half = lane >> 4;
#pragma unroll
  for (int t = 0; t < 4; ++t) {
#pragma unroll
    for (int r = 0; r < 8; ++r) {
      int m = m0 + wm * 16 + half * 8 + r;
      int n = n0 + wn * 64 + t * 16 + nl;
      if (m < M && n < N) {
        float v = acc[t][r] + (bias ? bias[m] : 0.0f);
        if (ACT == 1)      v = fmaxf(v, 0.0f);
        else if (ACT == 2) v = gelu_f(v);
        Y[(size_t)m * N + n] = v;
      }
    }
  }
}

// ---------------- S4 bidirectional state scan -------------------------------
// y_fwd[l] = 2 Re(sum_n coef0_n s_n[l]),  s = a*s + x   (l ascending)
// y_bwd[l] = 2 Re(sum_n coef1_n t_n[l]),  t[l-1] = x[l] + a*t[l]  (descending)
// One wave per (b,h,dir); lane = state n (32 conjugate pairs). dir=1 adds.
__global__ __launch_bounds__(256) void s4_scan(
    float* __restrict__ ybuf, const float* __restrict__ z,
    const float* __restrict__ log_dt,
    const float* __restrict__ A_re, const float* __restrict__ A_im,
    const float* __restrict__ B_re, const float* __restrict__ B_im,
    const float* __restrict__ C_re, const float* __restrict__ C_im,
    int H, int L, int dir)
{
  const int lane = threadIdx.x & 31;
  const int wave = threadIdx.x >> 5;
  const int h = blockIdx.x * 8 + wave;
  const int b = blockIdx.y;
  const int n = lane;

  float dt  = __expf(log_dt[h]);
  float ar  = -__expf(A_re[h * 32 + n]);
  float ai  = A_im[h * 32 + n];
  float dre = dt * ar, dimg = dt * ai;
  float em = __expf(dre);
  float er = em * __cosf(dimg), ei = em * __sinf(dimg);     // a = exp(dt*A)
  float nr = er - 1.0f, ni = ei;                            // exp(dtA)-1
  float den = ar * ar + ai * ai;
  float ir = ar / den, ii = -ai / den;                      // 1/A
  float qr = nr * ir - ni * ii, qi = nr * ii + ni * ir;
  float br = B_re[h * 32 + n], bi = B_im[h * 32 + n];
  float tr = br * qr - bi * qi, ti = br * qi + bi * qr;     // B*(e-1)/A
  const float* cR = C_re + ((size_t)dir * H + h) * 32;
  const float* cI = C_im + ((size_t)dir * H + h) * 32;
  float cr = cR[n] * tr - cI[n] * ti;
  float ci = cR[n] * ti + cI[n] * tr;                       // coef = C*...

  const float* zp = z    + ((size_t)b * H + h) * L;
  float*       yp = ybuf + ((size_t)b * H + h) * L;
  float sr = 0.0f, si = 0.0f;

  if (dir == 0) {
    for (int l = 0; l < L; ++l) {
      float xv  = zp[l];
      float nsr = er * sr - ei * si + xv;
      float nsi = er * si + ei * sr;
      sr = nsr; si = nsi;
      float c = 2.0f * (cr * sr - ci * si);
      c += __shfl_xor(c, 16, 32);
      c += __shfl_xor(c, 8, 32);
      c += __shfl_xor(c, 4, 32);
      c += __shfl_xor(c, 2, 32);
      c += __shfl_xor(c, 1, 32);
      if (n == 0) yp[l] = c;
    }
  } else {
    for (int l = L - 1; l >= 0; --l) {
      float c = 2.0f * (cr * sr - ci * si);
      c += __shfl_xor(c, 16, 32);
      c += __shfl_xor(c, 8, 32);
      c += __shfl_xor(c, 4, 32);
      c += __shfl_xor(c, 2, 32);
      c += __shfl_xor(c, 1, 32);
      if (n == 0) yp[l] += c;
      float xv  = zp[l];
      float nsr = er * sr - ei * si + xv;
      float nsi = er * si + ei * sr;
      sr = nsr; si = nsi;
    }
  }
}

// y = gelu(yscan + D[h]*z)
__global__ void s4_post(float* __restrict__ out, const float* __restrict__ y,
                        const float* __restrict__ z, const float* __restrict__ D,
                        int H, int L, int B)
{
  int idx = blockIdx.x * blockDim.x + threadIdx.x;
  int total = B * H * L;
  if (idx >= total) return;
  int h = (idx / L) % H;
  out[idx] = gelu_f(y[idx] + D[h] * z[idx]);
}

// GLU over channel split: out[b,h,l] = u2[b,h,l]*sigmoid(u2[b,H+h,l])
__global__ void glu_kernel(float* __restrict__ out, const float* __restrict__ u2,
                           int H, int L, int B)
{
  int idx = blockIdx.x * blockDim.x + threadIdx.x;
  int total = B * H * L;
  if (idx >= total) return;
  int b = idx / (H * L);
  int rem = idx - b * H * L;
  const float* ub = u2 + (size_t)b * 2 * H * L;
  float a = ub[rem];
  float g = ub[(size_t)H * L + rem];
  out[idx] = a * sigm_f(g);
}

// LayerNorm over channel axis (axis=1) of (B,H,L), eps=1e-3, optional +pt[b,h]
__global__ void ln_ch_kernel(float* __restrict__ out, const float* __restrict__ x,
                             const float* __restrict__ pt, const float* __restrict__ g,
                             const float* __restrict__ bta, int H, int L, int B)
{
  int idx = blockIdx.x * blockDim.x + threadIdx.x;
  int total = B * L;
  if (idx >= total) return;
  int b = idx / L, l = idx - b * L;
  const float* xb = x + (size_t)b * H * L + l;
  const float* pb = pt ? pt + (size_t)b * H : nullptr;
  float m = 0.0f;
  for (int h = 0; h < H; ++h) m += xb[(size_t)h * L] + (pb ? pb[h] : 0.0f);
  m /= (float)H;
  float v = 0.0f;
  for (int h = 0; h < H; ++h) {
    float d = xb[(size_t)h * L] + (pb ? pb[h] : 0.0f) - m;
    v += d * d;
  }
  v /= (float)H;
  float rs = rsqrtf(v + 1e-3f);
  float* ob = out + (size_t)b * H * L + l;
  for (int h = 0; h < H; ++h)
    ob[(size_t)h * L] = (xb[(size_t)h * L] + (pb ? pb[h] : 0.0f) - m) * rs * g[h] + bta[h];
}

// x += z + cond_conv(cond)  (kernel=stride=s patches over full SEQ cond)
__global__ void res_finish(float* __restrict__ x, const float* __restrict__ z,
                           const float* __restrict__ cond, const float* __restrict__ cw,
                           const float* __restrict__ cb, int H, int s, int L, int B)
{
  int idx = blockIdx.x * blockDim.x + threadIdx.x;
  int total = B * H * L;
  if (idx >= total) return;
  int b = idx / (H * L);
  int rem = idx - b * H * L;
  int h = rem / L, l = rem - h * L;
  float acc = cb[h];
  const float* cbase = cond + (size_t)b * 2 * SEQ;
  for (int c = 0; c < 2; ++c)
    for (int k = 0; k < s; ++k)
      acc += cw[(h * 2 + c) * s + k] * cbase[(size_t)c * SEQ + l * s + k];
  x[idx] = x[idx] + z[idx] + acc;
}

__global__ void add_kernel(float* __restrict__ x, const float* __restrict__ s, int n)
{
  int idx = blockIdx.x * blockDim.x + threadIdx.x;
  if (idx < n) x[idx] += s[idx];
}

// cond = concat(conditional*mask, mask) along channel
__global__ void build_cond(float* __restrict__ cond, const float* __restrict__ c,
                           const float* __restrict__ m, int n)
{
  int idx = blockIdx.x * blockDim.x + threadIdx.x;
  if (idx >= n) return;
  int b = idx / SEQ, l = idx - b * SEQ;
  cond[(size_t)b * 2 * SEQ + l]        = c[idx] * m[idx];
  cond[(size_t)b * 2 * SEQ + SEQ + l]  = m[idx];
}

// Diffusion-step sinusoidal embedding (B,128)
__global__ void dse_embed(float* __restrict__ out, const int* __restrict__ steps)
{
  int b = blockIdx.x, i = threadIdx.x;   // 64 threads
  float st = (float)steps[b];
  float f = __expf(-logf(10000.0f) * (float)i / 63.0f);
  out[b * 128 + i]      = __sinf(st * f);
  out[b * 128 + 64 + i] = __cosf(st * f);
}

// Small dense: out[b,o] = act(sum_i w[o,i]*in[b,i] + bias[o]); act 0=none 1=silu
__global__ void dense_small(float* __restrict__ out, const float* __restrict__ in,
                            const float* __restrict__ w, const float* __restrict__ bias,
                            int O, int I, int B, int act)
{
  int idx = blockIdx.x * blockDim.x + threadIdx.x;
  int total = B * O;
  if (idx >= total) return;
  int b = idx / O, o = idx - b * O;
  const float* ib = in + (size_t)b * I;
  const float* wr = w + (size_t)o * I;
  float acc = bias[o];
  for (int i = 0; i < I; ++i) acc += wr[i] * ib[i];
  if (act == 1) acc = silu_f(acc);
  out[idx] = acc;
}

// down: out[b, h*s+j, l'] = x[b, h, l'*s + j]   (B,H,L)->(B,H*s,L/s)
__global__ void down_reshape(float* __restrict__ out, const float* __restrict__ x,
                             int H, int s, int L, int B)
{
  int Ls = L / s;
  int idx = blockIdx.x * blockDim.x + threadIdx.x;
  int total = B * H * L;
  if (idx >= total) return;
  int b = idx / (H * s * Ls);
  int rem = idx - b * H * s * Ls;
  int hs = rem / Ls, lp = rem - hs * Ls;
  int h = hs / s, j = hs - h * s;
  out[idx] = x[((size_t)b * H + h) * L + lp * s + j];
}

// up: out[b, h, l*s + j] = y[b, h*s + j, l]   (B,HP,L)->(B,HP/s,L*s)
__global__ void up_reshape(float* __restrict__ out, const float* __restrict__ y,
                           int HP, int s, int L, int B)
{
  int Ho = HP / s, Lo = L * s;
  int idx = blockIdx.x * blockDim.x + threadIdx.x;
  int total = B * Ho * Lo;
  if (idx >= total) return;
  int b = idx / (Ho * Lo);
  int rem = idx - b * Ho * Lo;
  int h = rem / Lo, ll = rem - h * Lo;
  int l = ll / s, j = ll - l * s;
  out[idx] = y[((size_t)b * HP + h * s + j) * L + l];
}

// --------------------------- host-side orchestration ------------------------
namespace {

struct S4P { const float *A_im,*A_re,*B_im,*B_re,*C_im,*C_re,*D,*log_dt,*out_b,*out_w; };
struct FFP { const float *b1,*b2,*w1,*w2; };
struct Res {
  int isS4, H, s;
  const float *bta,*cond_b,*cond_w,*fc_t_b,*fc_t_w,*g;
  S4P s4; FFP ff;
};
struct Lin { const float *b,*w; };

// Walks jax tree-flatten order (dict keys sorted, lists in order).
// Supports two harness conventions: single concatenated params buffer
// (d_in[6], advance by element count) or one d_in entry per leaf.
struct PC {
  void* const* din;
  int idx;
  const float* flat;
  size_t off;
  bool per_leaf;
  const float* next(size_t n) {
    if (per_leaf) return (const float*)din[idx++];
    const float* p = flat + off; off += n; return p;
  }
};

static Res readRes(PC& pc, int isS4, int H, int s) {
  Res r{}; r.isS4 = isS4; r.H = H; r.s = s;
  // resp dict sorted: bta, cond_b, cond_w, fc_t_b, fc_t_w, [ff], g, [s4]
  r.bta    = pc.next(H);
  r.cond_b = pc.next(H);
  r.cond_w = pc.next((size_t)H * 2 * s);
  r.fc_t_b = pc.next(H);
  r.fc_t_w = pc.next((size_t)H * 512);
  if (!isS4) {
    r.ff.b1 = pc.next(2 * H);
    r.ff.b2 = pc.next(H);
    r.ff.w1 = pc.next((size_t)2 * H * H);
    r.ff.w2 = pc.next((size_t)H * 2 * H);
    r.g = pc.next(H);
  } else {
    r.g = pc.next(H);
    // s4 dict sorted: A_im A_re B_im B_re C_im C_re D log_dt out_b out_w
    r.s4.A_im = pc.next((size_t)H * 32);
    r.s4.A_re = pc.next((size_t)H * 32);
    r.s4.B_im = pc.next((size_t)H * 32);
    r.s4.B_re = pc.next((size_t)H * 32);
    r.s4.C_im = pc.next((size_t)2 * H * 32);
    r.s4.C_re = pc.next((size_t)2 * H * 32);
    r.s4.D      = pc.next(H);
    r.s4.log_dt = pc.next(H);
    r.s4.out_b  = pc.next(2 * H);
    r.s4.out_w  = pc.next((size_t)2 * H * H);
  }
  return r;
}

} // namespace

extern "C" void kernel_launch(void* const* d_in, const int* in_sizes, int n_in,
                              void* d_out, int out_size, void* d_ws, size_t ws_size,
                              hipStream_t stream)
{
  (void)in_sizes; (void)out_size; (void)ws_size;
  const float* noise       = (const float*)d_in[0];
  const float* conditional = (const float*)d_in[1];
  const float* maskp       = (const float*)d_in[2];
  const int*   steps       = (const int*)d_in[3];
  // d_in[4]=y_true, d_in[5]=loss_mask unused by forward.

  PC pc; pc.din = d_in; pc.idx = 6; pc.flat = (const float*)d_in[6];
  pc.off = 0; pc.per_leaf = (n_in > 7);

  // ----- parameter mapping in flatten order: c, d, f1, f2, init, norm, t1, t2, u
  Res cL[12];
  for (int i = 0; i < 6; ++i) { cL[2*i] = readRes(pc,1,256,4); cL[2*i+1] = readRes(pc,0,256,4); }
  Res dL[24]; Lin down0, down1;
  for (int i = 0; i < 6; ++i) { dL[2*i]    = readRes(pc,1,64,1);  dL[2*i+1]  = readRes(pc,0,64,1); }
  down0.b = pc.next(128); down0.w = pc.next((size_t)128*128);
  for (int i = 0; i < 6; ++i) { dL[12+2*i] = readRes(pc,1,128,2); dL[13+2*i] = readRes(pc,0,128,2); }
  down1.b = pc.next(256); down1.w = pc.next((size_t)256*256);
  const float* p_f1_b = pc.next(64);  const float* p_f1_w = pc.next((size_t)64*64);
  const float* p_f2_b = pc.next(1);   const float* p_f2_w = pc.next(64);
  const float* p_init_b = pc.next(64); const float* p_init_w = pc.next(64);
  const float* p_norm_b = pc.next(64); const float* p_norm_g = pc.next(64);
  const float* p_t1_b = pc.next(512); const float* p_t1_w = pc.next((size_t)512*128);
  const float* p_t2_b = pc.next(512); const float* p_t2_w = pc.next((size_t)512*512);
  Lin up0; up0.b = pc.next(256); up0.w = pc.next((size_t)256*256);
  Res u0L[12];
  for (int i = 0; i < 6; ++i) { u0L[2*i] = readRes(pc,1,128,2); u0L[2*i+1] = readRes(pc,0,128,2); }
  Lin up1; up1.b = pc.next(128); up1.w = pc.next((size_t)128*128);
  Res u1L[12];
  for (int i = 0; i < 6; ++i) { u1L[2*i] = readRes(pc,1,64,1);  u1L[2*i+1] = readRes(pc,0,64,1); }

  // ----- workspace layout (floats). All levels have B*H*L = 1,048,576.
  float* ws = (float*)d_ws;
  size_t off = 0;
  auto alloc = [&](size_t n) { float* p = ws + off; off += n; return p; };
  const size_t NE = (size_t)BATCH_N * 64 * SEQ;   // 1,048,576
  float* cond = alloc((size_t)BATCH_N * 2 * SEQ);
  float* dse0 = alloc((size_t)BATCH_N * 128);
  float* dse1 = alloc((size_t)BATCH_N * 512);
  float* dse2 = alloc((size_t)BATCH_N * 512);
  float* pt   = alloc((size_t)BATCH_N * 512);
  float* x    = alloc(NE);
  float* t0 = alloc(2 * NE);
  float* t1 = alloc(2 * NE);
  float* t2 = alloc(2 * NE);
  float* t3 = alloc(2 * NE);
  float* skip[27];
  for (int i = 0; i < 27; ++i) skip[i] = alloc(NE);

  auto gemm = [&](float* Y, const float* X, const float* W, const float* bsp,
                  int M, int N, int K, int act) {
    dim3 g((unsigned)((N + 127) / 128), (unsigned)((M + 63) / 64), (unsigned)BATCH_N);
    if (act == 0)      gemm_wmma_f16<0><<<g, 256, 0, stream>>>(Y, X, W, bsp, M, N, K);
    else if (act == 1) gemm_wmma_f16<1><<<g, 256, 0, stream>>>(Y, X, W, bsp, M, N, K);
    else               gemm_wmma_f16<2><<<g, 256, 0, stream>>>(Y, X, W, bsp, M, N, K);
  };

  auto run_res = [&](const Res& r, int L) {
    const int H = r.H;
    dense_small<<<dim3((unsigned)((BATCH_N*H + 255)/256)), 256, 0, stream>>>(
        pt, dse2, r.fc_t_w, r.fc_t_b, H, 512, BATCH_N, 0);
    ln_ch_kernel<<<dim3((unsigned)((BATCH_N*L + 255)/256)), 256, 0, stream>>>(
        t0, x, pt, r.g, r.bta, H, L, BATCH_N);
    const int n = BATCH_N * H * L;
    const float* zout;
    if (r.isS4) {
      s4_scan<<<dim3((unsigned)(H/8), (unsigned)BATCH_N), 256, 0, stream>>>(
          t1, t0, r.s4.log_dt, r.s4.A_re, r.s4.A_im, r.s4.B_re, r.s4.B_im,
          r.s4.C_re, r.s4.C_im, H, L, 0);
      s4_scan<<<dim3((unsigned)(H/8), (unsigned)BATCH_N), 256, 0, stream>>>(
          t1, t0, r.s4.log_dt, r.s4.A_re, r.s4.A_im, r.s4.B_re, r.s4.B_im,
          r.s4.C_re, r.s4.C_im, H, L, 1);
      s4_post<<<dim3((unsigned)((n + 255)/256)), 256, 0, stream>>>(t2, t1, t0, r.s4.D, H, L, BATCH_N);
      gemm(t3, t2, r.s4.out_w, r.s4.out_b, 2*H, L, H, 0);
      glu_kernel<<<dim3((unsigned)((n + 255)/256)), 256, 0, stream>>>(t1, t3, H, L, BATCH_N);
      zout = t1;
    } else {
      gemm(t1, t0, r.ff.w1, r.ff.b1, 2*H, L, H, 2);
      gemm(t2, t1, r.ff.w2, r.ff.b2, H, L, 2*H, 0);
      zout = t2;
    }
    res_finish<<<dim3((unsigned)((n + 255)/256)), 256, 0, stream>>>(
        x, zout, cond, r.cond_w, r.cond_b, H, r.s, L, BATCH_N);
  };

  auto save = [&](int i) {
    (void)hipMemcpyAsync(skip[i], x, NE * sizeof(float), hipMemcpyDeviceToDevice, stream);
  };
  auto addskip = [&](int i) {
    add_kernel<<<dim3((unsigned)((NE + 255)/256)), 256, 0, stream>>>(x, skip[i], (int)NE);
  };

  // ----- forward -----
  {
    int n = BATCH_N * SEQ;
    build_cond<<<dim3((unsigned)((n + 255)/256)), 256, 0, stream>>>(cond, conditional, maskp, n);
  }
  dse_embed<<<dim3((unsigned)BATCH_N), 64, 0, stream>>>(dse0, steps);
  dense_small<<<dim3((unsigned)((BATCH_N*512 + 255)/256)), 256, 0, stream>>>(
      dse1, dse0, p_t1_w, p_t1_b, 512, 128, BATCH_N, 1);
  dense_small<<<dim3((unsigned)((BATCH_N*512 + 255)/256)), 256, 0, stream>>>(
      dse2, dse1, p_t2_w, p_t2_b, 512, 512, BATCH_N, 1);

  gemm(x, noise, p_init_w, p_init_b, 64, SEQ, 1, 1);   // relu(init conv)

  save(0);
  int si = 1;
  for (int i = 0; i < 12; ++i) { run_res(dL[i], 8192); save(si++); }
  down_reshape<<<dim3((unsigned)((NE + 255)/256)), 256, 0, stream>>>(t0, x, 64, 2, 8192, BATCH_N);
  gemm(x, t0, down0.w, down0.b, 128, 4096, 128, 0);
  save(si++);
  for (int i = 12; i < 24; ++i) { run_res(dL[i], 4096); save(si++); }
  down_reshape<<<dim3((unsigned)((NE + 255)/256)), 256, 0, stream>>>(t0, x, 128, 2, 4096, BATCH_N);
  gemm(x, t0, down1.w, down1.b, 256, 2048, 256, 0);
  save(si++);   // si == 27

  for (int i = 0; i < 12; ++i) run_res(cL[i], 2048);
  addskip(--si);

  gemm(t0, x, up0.w, up0.b, 256, 2048, 256, 0);
  up_reshape<<<dim3((unsigned)((NE + 255)/256)), 256, 0, stream>>>(x, t0, 256, 2, 2048, BATCH_N);
  addskip(--si);
  for (int i = 0; i < 12; ++i) { run_res(u0L[i], 4096); addskip(--si); }

  gemm(t0, x, up1.w, up1.b, 128, 4096, 128, 0);
  up_reshape<<<dim3((unsigned)((NE + 255)/256)), 256, 0, stream>>>(x, t0, 128, 2, 4096, BATCH_N);
  addskip(--si);
  for (int i = 0; i < 12; ++i) { run_res(u1L[i], 8192); addskip(--si); }

  ln_ch_kernel<<<dim3((unsigned)((BATCH_N*SEQ + 255)/256)), 256, 0, stream>>>(
      t0, x, nullptr, p_norm_g, p_norm_b, 64, SEQ, BATCH_N);
  gemm(t1, t0, p_f1_w, p_f1_b, 64, SEQ, 64, 1);
  gemm((float*)d_out, t1, p_f2_w, p_f2_b, 1, SEQ, 64, 0);
}
